// SelfAttentive_5738076307851
// MI455X (gfx1250) — matrix operand order repaired and checked by reading, not compiled, not run on hardware
//
#include <hip/hip_runtime.h>

typedef __attribute__((ext_vector_type(16))) __bf16 v16bf;
typedef __attribute__((ext_vector_type(8)))  __bf16 v8bf;
typedef __attribute__((ext_vector_type(8)))  float  v8f;

#define B_   64
#define S_   1024
#define EP_  320
#define H_   256
#define NWG_ 16

__device__ __forceinline__ unsigned short f2bf(float f) {
  unsigned u = __float_as_uint(f);
  u += 0x7FFFu + ((u >> 16) & 1u);
  return (unsigned short)(u >> 16);
}

__device__ __forceinline__ float sigm(float x) { return 1.f / (1.f + __expf(-x)); }

// A-fragment (16x32 bf16, MxK): lane<16 holds row=lane, K {0..7}U{16..23}; lane>=16: row=lane-16, K {8..15}U{24..31}
__device__ __forceinline__ v16bf load_fragA(const unsigned short* base, int ld) {
  int lane = threadIdx.x & 31;
  const __bf16* p = (const __bf16*)(base + (size_t)(lane & 15) * ld + ((lane >> 4) << 3));
  v8bf lo = *(const v8bf*)(p);
  v8bf hi = *(const v8bf*)(p + 16);
  v16bf r;
#pragma unroll
  for (int i = 0; i < 8; ++i) { r[i] = lo[i]; r[i + 8] = hi[i]; }
  return r;
}

// B-fragment (32x16 bf16, KxN) sourced from W row-major (N rows x K cols): lane<16 K {0..7}U{16..23}; lane>=16 K {8..15}U{24..31}
__device__ __forceinline__ v16bf load_fragB(const unsigned short* base, int ld) {
  int lane = threadIdx.x & 31;
  const __bf16* p = (const __bf16*)(base + (size_t)(lane & 15) * ld + ((lane >> 4) << 3));
  v8bf lo = *(const v8bf*)(p);
  v8bf hi = *(const v8bf*)(p + 16);
  v16bf r;
#pragma unroll
  for (int i = 0; i < 8; ++i) { r[i] = lo[i]; r[i + 8] = hi[i]; }
  return r;
}

#define WMMA_BF16(a, b, c) \
  __builtin_amdgcn_wmma_f32_16x16x32_bf16(false, (a), false, (b), (short)0, (c), false, false)

// D-tile (16x16 f32) store: VGPR r -> row = r + 8*(lane>>4), col = lane&15
__device__ __forceinline__ void store_tile(float* tile, int ld, v8f a) {
  int lane = threadIdx.x & 31;
  float* p = tile + (size_t)((lane >> 4) << 3) * ld + (lane & 15);
#pragma unroll
  for (int r = 0; r < 8; ++r) p[r * ld] = a[r];
}

// -------- global step barrier (persistent scan kernel) --------
__device__ __forceinline__ void gbar(unsigned* ctr) {
  __syncthreads();
  if (threadIdx.x == 0) {
    __threadfence();
    atomicAdd(ctr, 1u);
    while (__hip_atomic_load(ctr, __ATOMIC_ACQUIRE, __HIP_MEMORY_SCOPE_AGENT) < NWG_)
      __builtin_amdgcn_s_sleep(1);
  }
  __syncthreads();
  __threadfence();
}

// ================= kernels =================

__global__ void k_init(unsigned* bar, int nbar, float* penal) {
  int i = blockIdx.x * blockDim.x + threadIdx.x;
  if (i < nbar) bar[i] = 0u;
  if (i == 0) *penal = 0.f;
}

__global__ void k_cvt(const float* __restrict__ src, unsigned short* __restrict__ dst,
                      int rows, int sc, int dc) {
  int i = blockIdx.x * blockDim.x + threadIdx.x;
  if (i >= rows * dc) return;
  int r = i / dc, c = i - r * dc;
  float v = (c < sc) ? src[(size_t)r * sc + c] : 0.f;
  dst[i] = f2bf(v);
}

__global__ void k_embed(const int* __restrict__ ids, const float* __restrict__ emb,
                        unsigned short* __restrict__ x16) {
  int row = blockIdx.x;                 // b*S + s
  int id = ids[row];
  const float* e = emb + (size_t)id * 300;
  unsigned short* o = x16 + (size_t)row * EP_;
  for (int c = threadIdx.x; c < EP_; c += blockDim.x)
    o[c] = f2bf(c < 300 ? e[c] : 0.f);
}

// XW = X(65536 x 320) @ W^T(320 x 1024), bf16 WMMA, f32 out. z selects direction.
__global__ void __launch_bounds__(128)
k_xw(const unsigned short* __restrict__ X, const unsigned short* __restrict__ Wf,
     const unsigned short* __restrict__ Wb, float* __restrict__ XWf, float* __restrict__ XWb) {
  const unsigned short* W = blockIdx.z ? Wb : Wf;
  float* XW = blockIdx.z ? XWb : XWf;
  int w = threadIdx.x >> 5;
  int row0 = blockIdx.x * 128 + w * 32;  // wave owns 2 M-tiles
  int col0 = blockIdx.y * 64;            // 4 N-tiles
  v8f acc[2][4];
#pragma unroll
  for (int i = 0; i < 2; ++i)
#pragma unroll
    for (int n = 0; n < 4; ++n)
#pragma unroll
      for (int e = 0; e < 8; ++e) acc[i][n][e] = 0.f;

  for (int kt = 0; kt < EP_ / 32; ++kt) {
    v16bf a0 = load_fragA(X + (size_t)row0 * EP_ + kt * 32, EP_);
    v16bf a1 = load_fragA(X + (size_t)(row0 + 16) * EP_ + kt * 32, EP_);
#pragma unroll
    for (int nt = 0; nt < 4; ++nt) {
      v16bf b = load_fragB(W + (size_t)(col0 + nt * 16) * EP_ + kt * 32, EP_);
      acc[0][nt] = WMMA_BF16(a0, b, acc[0][nt]);
      acc[1][nt] = WMMA_BF16(a1, b, acc[1][nt]);
    }
  }
#pragma unroll
  for (int i = 0; i < 2; ++i)
#pragma unroll
    for (int nt = 0; nt < 4; ++nt)
      store_tile(XW + (size_t)(row0 + i * 16) * 1024 + col0 + nt * 16, 1024, acc[i][nt]);
}

// Persistent LSTM scan. 16 WGs; WG g owns h columns [g*16, g*16+16) (=> gate cols g*16+gate*256).
// W_hh slice in LDS; h state as bf16 in global (L2-resident); device barrier each step.
__global__ void __launch_bounds__(128)
k_scan(int dir, const float* __restrict__ XW, const unsigned short* __restrict__ Whh16,
       const float* __restrict__ h0, const float* __restrict__ c0, const int* __restrict__ len,
       unsigned short* __restrict__ h16, float* __restrict__ Hout, unsigned short* __restrict__ Hout16,
       float* __restrict__ hrev0, float* __restrict__ outH, float* __restrict__ outC,
       unsigned* __restrict__ bar) {
  __shared__ unsigned short lds_w[4 * 16 * 256];   // [gate][16 rows][256 k]
  __shared__ int lds_len[B_];
  int g = blockIdx.x;
  int tid = threadIdx.x;
  int wave = tid >> 5, lane = tid & 31;

  if (tid < B_) lds_len[tid] = len[tid];
  // stage W_hh slice (rows gate*256 + g*16 + r) into LDS, 16B vectors
  for (int j = tid; j < 2048; j += 128) {
    int el = j * 8;
    int gate = el >> 12;
    int rem = el & 4095;
    int r = rem >> 8, c = rem & 255;
    int grow = gate * 256 + g * 16 + r;
    ((uint4*)lds_w)[j] = *(const uint4*)(Whh16 + (size_t)grow * 256 + c);
  }

  int col = g * 16 + (lane & 15);
  int rbase = wave * 16 + ((lane >> 4) << 3);
  float hreg[8], creg[8];
#pragma unroll
  for (int r = 0; r < 8; ++r) {
    int b = rbase + r;
    hreg[r] = h0[(size_t)dir * B_ * H_ + b * H_ + col];
    creg[r] = c0[(size_t)dir * B_ * H_ + b * H_ + col];
    h16[b * H_ + col] = f2bf(hreg[r]);
  }
  __syncthreads();
  gbar(&bar[0]);

  for (int t = 0; t < S_; ++t) {
    v8f acc[4];
#pragma unroll
    for (int gg = 0; gg < 4; ++gg)
#pragma unroll
      for (int e = 0; e < 8; ++e) acc[gg][e] = 0.f;

    for (int kt = 0; kt < H_ / 32; ++kt) {
      v16bf a = load_fragA(h16 + (size_t)(wave * 16) * H_ + kt * 32, H_);
#pragma unroll
      for (int gate = 0; gate < 4; ++gate) {
        v16bf b = load_fragB(lds_w + gate * 16 * 256 + kt * 32, 256);
        acc[gate] = WMMA_BF16(a, b, acc[gate]);
      }
    }

#pragma unroll
    for (int r = 0; r < 8; ++r) {
      int b = rbase + r;
      int lenb = lds_len[b];
      int tx = dir ? (lenb - 1 - t < 0 ? 0 : lenb - 1 - t) : t;
      size_t xbase = ((size_t)b * S_ + tx) * 1024;
      float gi = acc[0][r] + XW[xbase + 0 * H_ + col];
      float gf = acc[1][r] + XW[xbase + 1 * H_ + col];
      float gg = acc[2][r] + XW[xbase + 2 * H_ + col];
      float go = acc[3][r] + XW[xbase + 3 * H_ + col];
      float cn = sigm(gf) * creg[r] + sigm(gi) * tanhf(gg);
      float hn = sigm(go) * tanhf(cn);
      bool m = t < lenb;
      if (m) { creg[r] = cn; hreg[r] = hn; }
      h16[b * H_ + col] = f2bf(hreg[r]);
      if (dir == 0) {
        size_t ho = ((size_t)b * S_ + t) * 512 + col;
        Hout[ho] = hreg[r];
        Hout16[ho] = f2bf(hreg[r]);
      } else {
        if (m) {
          size_t ho = ((size_t)b * S_ + (lenb - 1 - t)) * 512 + 256 + col;
          Hout[ho] = hreg[r];
          Hout16[ho] = f2bf(hreg[r]);
        }
        if (t == 0) hrev0[b * H_ + col] = hreg[r];
      }
    }
    gbar(&bar[t + 1]);
  }

#pragma unroll
  for (int r = 0; r < 8; ++r) {
    int b = rbase + r;
    outH[(size_t)dir * B_ * H_ + b * H_ + col] = hreg[r];
    outC[(size_t)dir * B_ * H_ + b * H_ + col] = creg[r];
  }
}

// pad positions t >= len: backward half of Hout = h_rev0 (== out_b_rev[:,0])
__global__ void k_fill(const int* __restrict__ len, const float* __restrict__ hrev0,
                       float* __restrict__ Hout, unsigned short* __restrict__ Hout16) {
  int b = blockIdx.x;
  int l = len[b];
  int n = (S_ - l) * H_;
  for (int i = threadIdx.x; i < n; i += blockDim.x) {
    int t = l + i / H_, c = i - (i / H_) * H_;
    float v = hrev0[b * H_ + c];
    size_t o = ((size_t)b * S_ + t) * 512 + 256 + c;
    Hout[o] = v;
    Hout16[o] = f2bf(v);
  }
}

// scores[b][t][r] = mask ? (tanh(Hout@S1^T) @ S2^T) : -1e30 ; WMMA stage-1, VALU stage-2
__global__ void __launch_bounds__(32)
k_scores(const unsigned short* __restrict__ Hout16, const unsigned short* __restrict__ S116,
         const float* __restrict__ S2, const int* __restrict__ len, float* __restrict__ scores) {
  __shared__ float ta[16 * 64];
  int row0 = blockIdx.x * 16;
  int lane = threadIdx.x;
  v8f acc[4];
#pragma unroll
  for (int n = 0; n < 4; ++n)
#pragma unroll
    for (int e = 0; e < 8; ++e) acc[n][e] = 0.f;

  for (int kt = 0; kt < 512 / 32; ++kt) {
    v16bf a = load_fragA(Hout16 + (size_t)row0 * 512 + kt * 32, 512);
#pragma unroll
    for (int nt = 0; nt < 4; ++nt) {
      v16bf b = load_fragB(S116 + (size_t)(nt * 16) * 512 + kt * 32, 512);
      acc[nt] = WMMA_BF16(a, b, acc[nt]);
    }
  }
  int m0 = (lane >> 4) << 3, n = lane & 15;
#pragma unroll
  for (int nt = 0; nt < 4; ++nt)
#pragma unroll
    for (int r = 0; r < 8; ++r) ta[(m0 + r) * 64 + nt * 16 + n] = tanhf(acc[nt][r]);
  __syncthreads();

  for (int idx = lane; idx < 256; idx += 32) {
    int row = idx >> 4, rr = idx & 15;
    float s = 0.f;
    for (int d = 0; d < 64; ++d) s += ta[row * 64 + d] * S2[rr * 64 + d];
    int grow = row0 + row;
    int b = grow >> 10, t = grow & 1023;
    scores[(size_t)grow * 16 + rr] = (t < len[b]) ? s : -1e30f;
  }
}

__global__ void __launch_bounds__(256)
k_softmax(const float* __restrict__ scores, float* __restrict__ A) {
  __shared__ float red[256];
  int br = blockIdx.x;             // b*16 + r
  int b = br >> 4, r = br & 15;
  const float* s = scores + (size_t)b * S_ * 16 + r;  // stride 16
  int tid = threadIdx.x;
  float mx = -1e38f;
  for (int t = tid; t < S_; t += 256) mx = fmaxf(mx, s[(size_t)t * 16]);
  red[tid] = mx; __syncthreads();
  for (int o = 128; o > 0; o >>= 1) {
    if (tid < o) red[tid] = fmaxf(red[tid], red[tid + o]);
    __syncthreads();
  }
  mx = red[0]; __syncthreads();
  float sum = 0.f;
  for (int t = tid; t < S_; t += 256) sum += __expf(s[(size_t)t * 16] - mx);
  red[tid] = sum; __syncthreads();
  for (int o = 128; o > 0; o >>= 1) {
    if (tid < o) red[tid] += red[tid + o];
    __syncthreads();
  }
  float inv = 1.f / red[0];
  for (int t = tid; t < S_; t += 256)
    A[(size_t)br * S_ + t] = __expf(s[(size_t)t * 16] - mx) * inv;
}

// M[b][r][d] = sum_s A[b][r][s] * Hout[b][s][d]  (streams Hout once per batch)
__global__ void __launch_bounds__(512)
k_attnM(const float* __restrict__ A, const float* __restrict__ Hout, float* __restrict__ Mout) {
  __shared__ float la[16 * 64];
  int b = blockIdx.x, d = threadIdx.x;
  float acc[16];
#pragma unroll
  for (int r = 0; r < 16; ++r) acc[r] = 0.f;
  for (int s0 = 0; s0 < S_; s0 += 64) {
    __syncthreads();
    for (int i = threadIdx.x; i < 1024; i += 512) {
      int r = i >> 6, sc = i & 63;
      la[i] = A[((size_t)b * 16 + r) * S_ + s0 + sc];
    }
    __syncthreads();
    for (int sc = 0; sc < 64; ++sc) {
      float hv = Hout[((size_t)b * S_ + s0 + sc) * 512 + d];
#pragma unroll
      for (int r = 0; r < 16; ++r) acc[r] += la[r * 64 + sc] * hv;
    }
  }
#pragma unroll
  for (int r = 0; r < 16; ++r) Mout[((size_t)b * 16 + r) * 512 + d] = acc[r];
}

__global__ void __launch_bounds__(256)
k_aat(const float* __restrict__ A, float* __restrict__ penal) {
  __shared__ float red[256];
  int b = blockIdx.x, tid = threadIdx.x;
  int r = tid >> 4, q = tid & 15;
  const float* ar = A + ((size_t)b * 16 + r) * S_;
  const float* aq = A + ((size_t)b * 16 + q) * S_;
  float s = 0.f;
  for (int t = 0; t < S_; ++t) s += ar[t] * aq[t];
  float d = s - (r == q ? 1.f : 0.f);
  red[tid] = d * d; __syncthreads();
  for (int o = 128; o > 0; o >>= 1) {
    if (tid < o) red[tid] += red[tid + o];
    __syncthreads();
  }
  if (tid == 0) atomicAdd(penal, red[0] * (1.f / 64.f));
}

__global__ void __launch_bounds__(256)
k_head(const float* __restrict__ BM, const float* __restrict__ W, const float* __restrict__ bias,
       const float* __restrict__ Wd, const float* __restrict__ bd, int nout, float* __restrict__ out) {
  __shared__ float bm[8192];
  __shared__ float hid[512];
  int b = blockIdx.x;
  for (int i = threadIdx.x; i < 8192; i += 256) bm[i] = BM[(size_t)b * 8192 + i];
  __syncthreads();
  for (int mh = threadIdx.x; mh < 512; mh += 256) {
    const float* w = W + (size_t)mh * 8192;
    float s = bias[mh];
    for (int k = 0; k < 8192; ++k) s += w[k] * bm[k];
    hid[mh] = fmaxf(s, 0.f);
  }
  __syncthreads();
  if (threadIdx.x < nout) {
    const float* w = Wd + (size_t)threadIdx.x * 512;
    float s = bd[threadIdx.x];
    for (int k = 0; k < 512; ++k) s += w[k] * hid[k];
    out[(size_t)b * nout + threadIdx.x] = s;
  }
}

// ================= launch =================

extern "C" void kernel_launch(void* const* d_in, const int* in_sizes, int n_in,
                              void* d_out, int out_size, void* d_ws, size_t ws_size,
                              hipStream_t stream) {
  const int*   ids   = (const int*)d_in[0];
  const int*   len   = (const int*)d_in[1];
  const float* h0    = (const float*)d_in[2];
  const float* c0    = (const float*)d_in[3];
  const float* emb   = (const float*)d_in[4];
  const float* Wihf  = (const float*)d_in[5];
  const float* Whhf  = (const float*)d_in[6];
  const float* Wihb  = (const float*)d_in[7];
  const float* Whhb  = (const float*)d_in[8];
  const float* S1    = (const float*)d_in[9];
  const float* S2    = (const float*)d_in[10];
  const float* Wet   = (const float*)d_in[11];
  const float* bet   = (const float*)d_in[12];
  const float* Wdet  = (const float*)d_in[13];
  const float* bdet  = (const float*)d_in[14];
  const float* Wprt  = (const float*)d_in[15];
  const float* bprt  = (const float*)d_in[16];
  const float* Wdprt = (const float*)d_in[17];
  const float* bdprt = (const float*)d_in[18];
  const float* Wpop  = (const float*)d_in[19];
  const float* bpop  = (const float*)d_in[20];
  const float* Wdpop = (const float*)d_in[21];
  const float* bdpop = (const float*)d_in[22];
  float* out = (float*)d_out;

  char* w = (char*)d_ws;
  size_t off = 0;
  auto take = [&](size_t bytes) -> char* {
    char* p = w + off;
    off += (bytes + 255) & ~(size_t)255;
    return p;
  };
  unsigned short* x16    = (unsigned short*)take((size_t)B_ * S_ * EP_ * 2);
  unsigned short* wihf16 = (unsigned short*)take((size_t)1024 * EP_ * 2);
  unsigned short* wihb16 = (unsigned short*)take((size_t)1024 * EP_ * 2);
  unsigned short* whhf16 = (unsigned short*)take((size_t)1024 * H_ * 2);
  unsigned short* whhb16 = (unsigned short*)take((size_t)1024 * H_ * 2);
  unsigned short* s116   = (unsigned short*)take((size_t)64 * 512 * 2);
  float* xwf             = (float*)take((size_t)B_ * S_ * 1024 * 4);
  float* xwb             = (float*)take((size_t)B_ * S_ * 1024 * 4);
  float* hout            = (float*)take((size_t)B_ * S_ * 512 * 4);
  unsigned short* hout16 = (unsigned short*)take((size_t)B_ * S_ * 512 * 2);
  float* scores          = (float*)take((size_t)B_ * S_ * 16 * 4);
  unsigned short* h16f   = (unsigned short*)take((size_t)B_ * H_ * 2);
  unsigned short* h16b   = (unsigned short*)take((size_t)B_ * H_ * 2);
  float* hrev0           = (float*)take((size_t)B_ * H_ * 4);
  float* Mbuf            = (float*)take((size_t)B_ * 16 * 512 * 4);
  unsigned* bar          = (unsigned*)take((size_t)2 * (S_ + 1) * 4);

  float* o_et    = out;
  float* o_prt   = out + 3200;
  float* o_popt  = out + 3968;
  float* o_h     = out + 4480;
  float* o_c     = out + 37248;
  float* o_penal = out + 70016;
  float* o_A     = out + 70017;

  k_init<<<9, 256, 0, stream>>>(bar, 2 * (S_ + 1), o_penal);

  k_cvt<<<(1024 * EP_ + 255) / 256, 256, 0, stream>>>(Wihf, wihf16, 1024, 300, EP_);
  k_cvt<<<(1024 * EP_ + 255) / 256, 256, 0, stream>>>(Wihb, wihb16, 1024, 300, EP_);
  k_cvt<<<(1024 * H_ + 255) / 256, 256, 0, stream>>>(Whhf, whhf16, 1024, H_, H_);
  k_cvt<<<(1024 * H_ + 255) / 256, 256, 0, stream>>>(Whhb, whhb16, 1024, H_, H_);
  k_cvt<<<(64 * 512 + 255) / 256, 256, 0, stream>>>(S1, s116, 64, 512, 512);

  k_embed<<<B_ * S_, 128, 0, stream>>>(ids, emb, x16);

  k_xw<<<dim3((B_ * S_) / 128, 1024 / 64, 2), 128, 0, stream>>>(x16, wihf16, wihb16, xwf, xwb);

  k_scan<<<NWG_, 128, 0, stream>>>(0, xwf, whhf16, h0, c0, len, h16f, hout, hout16,
                                   hrev0, o_h, o_c, bar);
  k_scan<<<NWG_, 128, 0, stream>>>(1, xwb, whhb16, h0, c0, len, h16b, hout, hout16,
                                   hrev0, o_h, o_c, bar + (S_ + 1));

  k_fill<<<B_, 256, 0, stream>>>(len, hrev0, hout, hout16);

  k_scores<<<(B_ * S_) / 16, 32, 0, stream>>>(hout16, s116, S2, len, scores);
  k_softmax<<<B_ * 16, 256, 0, stream>>>(scores, o_A);
  k_attnM<<<B_, 512, 0, stream>>>(o_A, hout, Mbuf);
  k_aat<<<B_, 256, 0, stream>>>(o_A, o_penal);

  k_head<<<B_, 256, 0, stream>>>(Mbuf, Wet, bet, Wdet, bdet, 50, o_et);
  k_head<<<B_, 256, 0, stream>>>(Mbuf, Wprt, bprt, Wdprt, bdprt, 12, o_prt);
  k_head<<<B_, 256, 0, stream>>>(Mbuf, Wpop, bpop, Wdpop, bdpop, 8, o_popt);
}
